// JointAttention_66589172957751
// MI455X (gfx1250) — matrix operand order repaired and checked by reading, not compiled, not run on hardware
//
#include <hip/hip_runtime.h>
#include <hip/hip_bf16.h>
#include <hip/hip_fp16.h>

#define DIMN 3072
#define NH   24
#define HDD  128
#define TXTN 256
#define IMGN 2048
#define SEQN (TXTN + IMGN)   // 2304

typedef __attribute__((ext_vector_type(16))) _Float16 v16h;
typedef __attribute__((ext_vector_type(8)))  _Float16 v8h;
typedef __attribute__((ext_vector_type(8)))  float    v8f;
typedef __attribute__((ext_vector_type(4)))  unsigned int u32x4;
typedef __attribute__((ext_vector_type(8)))  int      i32x8;
typedef __attribute__((ext_vector_type(4)))  int      i32x4;

// ---------------------------------------------------------------------------
// WMMA fragment helpers (CDNA5 wave32, V_WMMA_F32_16X16X32_F16)
//
// A fragment (16x32): lane<16: row M=lane,    halves 0..7=K[0..7],  8..15=K[16..23]
//                     lane>=16: row M=lane-16, halves 0..7=K[8..15], 8..15=K[24..31]
// B fragment (32x16; source N x K row-major): lane<16: col N=lane, halves 0..7=K[0..7],  8..15=K[16..23]
//                     lane>=16: col N=lane-16, halves 0..7=K[8..15], 8..15=K[24..31]
// ---------------------------------------------------------------------------
static __device__ __forceinline__ v16h load_a_frag(const _Float16* base, int ld,
                                                   int row0, int k0, int lane) {
  int r  = lane & 15;
  int hi = lane >> 4;
  const _Float16* p = base + (size_t)(row0 + r) * ld + k0 + hi * 8;
  v8h lo  = *(const v8h*)(p);
  v8h hi8 = *(const v8h*)(p + 16);
  return __builtin_shufflevector(lo, hi8, 0,1,2,3,4,5,6,7,8,9,10,11,12,13,14,15);
}

static __device__ __forceinline__ v16h load_b_frag(const _Float16* base, int ld,
                                                   int n0, int k0, int lane) {
  int c  = lane & 15;
  int hi = lane >> 4;
  const _Float16* p = base + (size_t)(n0 + c) * ld + k0 + hi * 8;
  v8h lo  = *(const v8h*)(p);
  v8h hi8 = *(const v8h*)(p + 16);
  return __builtin_shufflevector(lo, hi8, 0,1,2,3,4,5,6,7,8,9,10,11,12,13,14,15);
}

// LDS tiles are written by the TDM with an 80B row pitch (40 halves):
// pad_interval = 16 DWORDs (one 32-elem f16 row), pad_amount = 4 DWORDs.
#define LDS_PITCH 40

static __device__ __forceinline__ v16h lds_a_frag(const _Float16* t, int row0, int lane) {
  int r  = lane & 15;
  int hi = lane >> 4;
  const _Float16* p = t + (row0 + r) * LDS_PITCH + hi * 8;
  v8h lo  = *(const v8h*)(p);
  v8h hi8 = *(const v8h*)(p + 16);
  return __builtin_shufflevector(lo, hi8, 0,1,2,3,4,5,6,7,8,9,10,11,12,13,14,15);
}

static __device__ __forceinline__ v16h lds_b_frag(const _Float16* t, int col0, int lane) {
  int c  = lane & 15;
  int hi = lane >> 4;
  const _Float16* p = t + (col0 + c) * LDS_PITCH + hi * 8;
  v8h lo  = *(const v8h*)(p);
  v8h hi8 = *(const v8h*)(p + 16);
  return __builtin_shufflevector(lo, hi8, 0,1,2,3,4,5,6,7,8,9,10,11,12,13,14,15);
}

static __device__ __forceinline__ v8f wmma_f16(v16h a, v16h b, v8f c) {
  return __builtin_amdgcn_wmma_f32_16x16x32_f16(false, a, false, b,
                                                (short)0, c, false, false);
}

// ---------------------------------------------------------------------------
// Tensor Data Mover: 2D tile load Global -> LDS (D# per CDNA5 ISA ch.8)
//   group0: [1:0]=count=1, [63:32]=lds_addr, [120:64]=global_addr, [127:126]=type=2
//   group1: data_size=2B, pad_enable (16DW interval, 4DW pad), tensor/tile dims,
//           tensor_dim0_stride = row pitch in elements.
// ---------------------------------------------------------------------------
static __device__ __forceinline__ void tdm_load_tile_2d(
    const _Float16* gptr,       // global address of tile start
    unsigned lds_off,           // LDS byte offset of tile destination
    int tensor_w, int tensor_h, // tensor extent (elements / rows)
    int tile_w, int tile_h,     // tile extent (elements / rows)
    int stride_elems) {         // row stride in elements
  unsigned long long ga = (unsigned long long)(uintptr_t)gptr;
  u32x4 g0;
  g0[0] = 1u;                                               // count=1 (valid)
  g0[1] = lds_off;
  g0[2] = (unsigned)(ga & 0xFFFFFFFFu);
  g0[3] = (unsigned)((ga >> 32) & 0x01FFFFFFu) | (2u << 30); // type=2 ("image")
  i32x8 g1;
  g1[0] = (1 << 16)                 // data_size = 1 -> 2 bytes
        | (1 << 20)                 // pad_enable
        | (3 << 22)                 // pad_interval: 16 DWORDs
        | (3 << 25);                // pad_amount: 4 DWORDs (16B per row)
  g1[1] = (tensor_w & 0xFFFF) << 16;
  g1[2] = ((unsigned)tensor_w >> 16) | ((tensor_h & 0xFFFF) << 16);
  g1[3] = ((unsigned)tensor_h >> 16) | (tile_w << 16);
  g1[4] = tile_h & 0xFFFF;          // tile_dim1; tile_dim2 = 0
  g1[5] = stride_elems;             // tensor_dim0_stride[31:0]
  g1[6] = 0;
  g1[7] = 0;
  i32x4 gz = {0, 0, 0, 0};
#if defined(__clang_major__) && __clang_major__ >= 23
  i32x8 gz8 = {0, 0, 0, 0, 0, 0, 0, 0};
  __builtin_amdgcn_tensor_load_to_lds(g0, g1, gz, gz, gz8, 0);
#else
  __builtin_amdgcn_tensor_load_to_lds(g0, g1, gz, gz, 0);
#endif
}

// ---------------------------------------------------------------------------
// fp32 -> f16 cast
// ---------------------------------------------------------------------------
__global__ void cast_f32_to_f16(const float* __restrict__ src,
                                _Float16* __restrict__ dst, int n) {
  int i = (blockIdx.x * blockDim.x + threadIdx.x) * 4;
  if (i + 3 < n) {
    float4 v = *(const float4*)(src + i);
    dst[i + 0] = (_Float16)v.x;
    dst[i + 1] = (_Float16)v.y;
    dst[i + 2] = (_Float16)v.z;
    dst[i + 3] = (_Float16)v.w;
  } else {
    for (; i < n; ++i) dst[i] = (_Float16)src[i];
  }
}

// ---------------------------------------------------------------------------
// C[M x N] = A[M x K] * W[N x K]^T + bias[N]   (f16 in, f32 out)
// block = 256 threads = 8 waves; block tile 64(M) x 128(N); wave tile 32x32.
// A/B tiles DMA'd to LDS by the Tensor Data Mover, double buffered
// (wave0 -> A tile, wave1 -> B tile), tracked with TENSORcnt.
// ---------------------------------------------------------------------------
__global__ __launch_bounds__(256) void gemm_wmma_tdm(
    const _Float16* __restrict__ A,
    const _Float16* __restrict__ W,
    const float*    __restrict__ bias,
    float*          __restrict__ C,
    int M, int N, int K) {
  __shared__ __align__(16) _Float16 lA[2][64][LDS_PITCH];    // 2 x 5 KB
  __shared__ __align__(16) _Float16 lB[2][128][LDS_PITCH];   // 2 x 10 KB

  int lane = threadIdx.x & 31;
  int wave = threadIdx.x >> 5;
  int wy = wave >> 2;                    // 0..1
  int wx = wave & 3;                     // 0..3
  int rowB = blockIdx.y * 64;
  int colB = blockIdx.x * 128;

  const int TK  = 32;
  const int nkt = K / TK;

  auto issue = [&](int kt, int buf) {
    if (wave == 0) {
      tdm_load_tile_2d(A + (size_t)rowB * K + kt * TK,
                       (unsigned)(uintptr_t)&lA[buf][0][0],
                       K, M, TK, 64, K);
    } else if (wave == 1) {
      tdm_load_tile_2d(W + (size_t)colB * K + kt * TK,
                       (unsigned)(uintptr_t)&lB[buf][0][0],
                       K, N, TK, 128, K);
    }
  };

  issue(0, 0);

  v8f acc00 = {}, acc01 = {}, acc10 = {}, acc11 = {};

  for (int kt = 0; kt < nkt; ++kt) {
    int buf = kt & 1;
    if (kt + 1 < nkt) issue(kt + 1, buf ^ 1);    // prefetch next tile via TDM
    if (wave < 2) {
      if (kt + 1 < nkt) __builtin_amdgcn_s_wait_tensorcnt(1);
      else              __builtin_amdgcn_s_wait_tensorcnt(0);
    }
    __syncthreads();                              // tile[buf] visible to all waves

    v16h a0 = lds_a_frag(&lA[buf][0][0], wy * 32,      lane);
    v16h a1 = lds_a_frag(&lA[buf][0][0], wy * 32 + 16, lane);
    v16h b0 = lds_b_frag(&lB[buf][0][0], wx * 32,      lane);
    v16h b1 = lds_b_frag(&lB[buf][0][0], wx * 32 + 16, lane);
    acc00 = wmma_f16(a0, b0, acc00);
    acc01 = wmma_f16(a0, b1, acc01);
    acc10 = wmma_f16(a1, b0, acc10);
    acc11 = wmma_f16(a1, b1, acc11);

    __syncthreads();                              // reads done before buf reused
  }

  int c  = lane & 15;
  int hi = lane >> 4;
  int row0 = rowB + wy * 32;
  int col0 = colB + wx * 32;
  float bi0 = bias[col0 + c];
  float bi1 = bias[col0 + 16 + c];
#pragma unroll
  for (int r = 0; r < 8; ++r) {
    int row = row0 + r + 8 * hi;
    C[(size_t)row        * N + col0 + c]      = acc00[r] + bi0;
    C[(size_t)row        * N + col0 + 16 + c] = acc01[r] + bi1;
    C[(size_t)(row + 16) * N + col0 + c]      = acc10[r] + bi0;
    C[(size_t)(row + 16) * N + col0 + 16 + c] = acc11[r] + bi1;
  }
}

// ---------------------------------------------------------------------------
// RoPE on q,k (fp32 in) -> f16 packed q,k.  Sequence layout: [txt | img].
// ---------------------------------------------------------------------------
__global__ void rope_pack(const float* __restrict__ qf, const float* __restrict__ kf,
                          const float* __restrict__ img_cos, const float* __restrict__ img_sin,
                          const float* __restrict__ txt_cos, const float* __restrict__ txt_sin,
                          _Float16* __restrict__ qh, _Float16* __restrict__ kh) {
  int idx = blockIdx.x * blockDim.x + threadIdx.x;     // SEQN * DIMN/2
  if (idx >= SEQN * (DIMN / 2)) return;
  int s  = idx / (DIMN / 2);
  int pr = idx % (DIMN / 2);
  int h  = pr / (HDD / 2);
  int dp = pr % (HDD / 2);
  int d0 = 2 * dp, d1 = d0 + 1;
  const float* cs; const float* sn; int pos;
  if (s < TXTN) { cs = txt_cos; sn = txt_sin; pos = s; }
  else          { cs = img_cos; sn = img_sin; pos = s - TXTN; }
  float c0 = cs[pos * HDD + d0], c1 = cs[pos * HDD + d1];
  float s0 = sn[pos * HDD + d0], s1 = sn[pos * HDD + d1];
  size_t o = (size_t)s * DIMN + h * HDD;
  float q0 = qf[o + d0], q1 = qf[o + d1];
  qh[o + d0] = (_Float16)(q0 * c0 - q1 * s0);
  qh[o + d1] = (_Float16)(q1 * c1 + q0 * s1);
  float k0 = kf[o + d0], k1 = kf[o + d1];
  kh[o + d0] = (_Float16)(k0 * c0 - k1 * s0);
  kh[o + d1] = (_Float16)(k1 * c1 + k0 * s1);
}

// vT[d][s] = v[s][d]  (f32 -> f16 transpose so PV B-fragments are contiguous)
__global__ void pack_vT(const float* __restrict__ vf, _Float16* __restrict__ vT) {
  int idx = blockIdx.x * blockDim.x + threadIdx.x;     // DIMN * SEQN
  if (idx >= DIMN * SEQN) return;
  int d = idx / SEQN;
  int s = idx % SEQN;
  vT[idx] = (_Float16)vf[(size_t)s * DIMN + d];
}

// ---------------------------------------------------------------------------
// Flash attention: grid (NH, SEQN/128), block 256 = 8 waves.
// Each wave: 16 queries x full head-dim 128.
// ---------------------------------------------------------------------------
__global__ __launch_bounds__(256) void attn_wmma(
    const _Float16* __restrict__ qh,
    const _Float16* __restrict__ kh,
    const _Float16* __restrict__ vT,
    _Float16* __restrict__ ao) {
  __shared__ __align__(16) _Float16 plds[8][16][32];   // per-wave private P tile

  int lane = threadIdx.x & 31;
  int wave = threadIdx.x >> 5;
  int head = blockIdx.x;
  int q0   = blockIdx.y * 128 + wave * 16;
  int dbase = head * HDD;
  const float scale = 0.08838834764831845f;            // 1/sqrt(128)

  v16h qfr[4];
#pragma unroll
  for (int i = 0; i < 4; ++i)
    qfr[i] = load_a_frag(qh, DIMN, q0, dbase + 32 * i, lane);

  v8f o[8];
#pragma unroll
  for (int j = 0; j < 8; ++j) o[j] = (v8f){};
  float m[8], l[8];
#pragma unroll
  for (int r = 0; r < 8; ++r) { m[r] = -1e30f; l[r] = 0.f; }

  int c  = lane & 15;
  int hi = lane >> 4;

  for (int kb = 0; kb < SEQN; kb += 32) {
    // ---- scores: S[16q x 32k] = Q * K^T (two 16-wide N tiles) ----
    v8f s0 = {}, s1 = {};
#pragma unroll
    for (int i = 0; i < 4; ++i) {
      v16h b0 = load_b_frag(kh, DIMN, kb,      dbase + 32 * i, lane);
      v16h b1 = load_b_frag(kh, DIMN, kb + 16, dbase + 32 * i, lane);
      s0 = wmma_f16(qfr[i], b0, s0);
      s1 = wmma_f16(qfr[i], b1, s1);
    }

    // ---- online softmax (row stats across 16-lane half-waves) ----
    float p0[8], p1[8], alpha[8];
#pragma unroll
    for (int r = 0; r < 8; ++r) {
      float a0 = s0[r] * scale;
      float a1 = s1[r] * scale;
      float t = fmaxf(a0, a1);
      t = fmaxf(t, __shfl_xor(t, 1, 32));
      t = fmaxf(t, __shfl_xor(t, 2, 32));
      t = fmaxf(t, __shfl_xor(t, 4, 32));
      t = fmaxf(t, __shfl_xor(t, 8, 32));
      float mn = fmaxf(m[r], t);
      alpha[r] = __expf(m[r] - mn);
      m[r] = mn;
      p0[r] = __expf(a0 - mn);
      p1[r] = __expf(a1 - mn);
      float rs = p0[r] + p1[r];
      rs += __shfl_xor(rs, 1, 32);
      rs += __shfl_xor(rs, 2, 32);
      rs += __shfl_xor(rs, 4, 32);
      rs += __shfl_xor(rs, 8, 32);
      l[r] = l[r] * alpha[r] + rs;
    }
#pragma unroll
    for (int j = 0; j < 8; ++j)
#pragma unroll
      for (int r = 0; r < 8; ++r) o[j][r] *= alpha[r];

    // ---- stage P: C-layout -> row-major LDS -> A-fragment layout ----
#pragma unroll
    for (int r = 0; r < 8; ++r) {
      plds[wave][r + 8 * hi][c]      = (_Float16)p0[r];
      plds[wave][r + 8 * hi][c + 16] = (_Float16)p1[r];
    }
    asm volatile("s_wait_dscnt 0" ::: "memory");       // wave-private region, no barrier
    int r2 = lane & 15;
    v8h lo  = *(const v8h*)&plds[wave][r2][hi * 8];
    v8h hi8 = *(const v8h*)&plds[wave][r2][hi * 8 + 16];
    v16h pa = __builtin_shufflevector(lo, hi8,
              0,1,2,3,4,5,6,7,8,9,10,11,12,13,14,15);

    // ---- O += P * V   (contraction over 32 keys, 8 d-tiles) ----
#pragma unroll
    for (int j = 0; j < 8; ++j) {
      v16h bv = load_b_frag(vT, SEQN, dbase + 16 * j, kb, lane);
      o[j] = wmma_f16(pa, bv, o[j]);
    }
  }

  // ---- normalize + write heads-concat layout ao[s][h*128+d] ----
#pragma unroll
  for (int j = 0; j < 8; ++j) {
#pragma unroll
    for (int r = 0; r < 8; ++r) {
      int row = q0 + r + 8 * hi;
      int col = dbase + 16 * j + c;
      ao[(size_t)row * DIMN + col] = (_Float16)(o[j][r] / l[r]);
    }
  }
}

// ---------------------------------------------------------------------------
extern "C" void kernel_launch(void* const* d_in, const int* in_sizes, int n_in,
                              void* d_out, int out_size, void* d_ws, size_t ws_size,
                              hipStream_t stream) {
  const float* img_x   = (const float*)d_in[0];
  const float* txt_x   = (const float*)d_in[1];
  const float* img_cos = (const float*)d_in[2];
  const float* img_sin = (const float*)d_in[3];
  const float* txt_cos = (const float*)d_in[4];
  const float* txt_sin = (const float*)d_in[5];
  const float* Wq  = (const float*)d_in[6];  const float* bq  = (const float*)d_in[7];
  const float* Wk  = (const float*)d_in[8];  const float* bk  = (const float*)d_in[9];
  const float* Wv  = (const float*)d_in[10]; const float* bv  = (const float*)d_in[11];
  const float* Waq = (const float*)d_in[12]; const float* baq = (const float*)d_in[13];
  const float* Wak = (const float*)d_in[14]; const float* bak = (const float*)d_in[15];
  const float* Wav = (const float*)d_in[16]; const float* bav = (const float*)d_in[17];
  const float* Wo  = (const float*)d_in[18]; const float* bo  = (const float*)d_in[19];
  const float* Wao = (const float*)d_in[20]; const float* bao = (const float*)d_in[21];
  float* out = (float*)d_out;                // [img_out 2048x3072 | txt_out 256x3072]

  // ---- workspace carve (~307 MB) ----
  char* p = (char*)d_ws;
  auto take = [&](size_t bytes) { char* r = p; p += (bytes + 255) & ~(size_t)255; return (void*)r; };
  const size_t WSZ = (size_t)DIMN * DIMN;
  const size_t ASZ = (size_t)SEQN * DIMN;
  _Float16* Wq_h  = (_Float16*)take(WSZ * 2);
  _Float16* Wk_h  = (_Float16*)take(WSZ * 2);
  _Float16* Wv_h  = (_Float16*)take(WSZ * 2);
  _Float16* Waq_h = (_Float16*)take(WSZ * 2);
  _Float16* Wak_h = (_Float16*)take(WSZ * 2);
  _Float16* Wav_h = (_Float16*)take(WSZ * 2);
  _Float16* Wo_h  = (_Float16*)take(WSZ * 2);
  _Float16* Wao_h = (_Float16*)take(WSZ * 2);
  _Float16* X_h   = (_Float16*)take(ASZ * 2);
  float*    qf    = (float*)   take(ASZ * 4);
  float*    kf    = (float*)   take(ASZ * 4);
  float*    vf    = (float*)   take(ASZ * 4);
  _Float16* qh    = (_Float16*)take(ASZ * 2);
  _Float16* kh    = (_Float16*)take(ASZ * 2);
  _Float16* vT    = (_Float16*)take(ASZ * 2);
  _Float16* ao    = (_Float16*)take(ASZ * 2);

  // ---- casts ----
  {
    int n = (int)WSZ, g = (n / 4 + 255) / 256;
    cast_f32_to_f16<<<g, 256, 0, stream>>>(Wq,  Wq_h,  n);
    cast_f32_to_f16<<<g, 256, 0, stream>>>(Wk,  Wk_h,  n);
    cast_f32_to_f16<<<g, 256, 0, stream>>>(Wv,  Wv_h,  n);
    cast_f32_to_f16<<<g, 256, 0, stream>>>(Waq, Waq_h, n);
    cast_f32_to_f16<<<g, 256, 0, stream>>>(Wak, Wak_h, n);
    cast_f32_to_f16<<<g, 256, 0, stream>>>(Wav, Wav_h, n);
    cast_f32_to_f16<<<g, 256, 0, stream>>>(Wo,  Wo_h,  n);
    cast_f32_to_f16<<<g, 256, 0, stream>>>(Wao, Wao_h, n);
    int nt = TXTN * DIMN, ni = IMGN * DIMN;
    cast_f32_to_f16<<<(nt / 4 + 255) / 256, 256, 0, stream>>>(txt_x, X_h, nt);
    cast_f32_to_f16<<<(ni / 4 + 255) / 256, 256, 0, stream>>>(img_x, X_h + (size_t)TXTN * DIMN, ni);
  }

  // ---- q/k/v projections (txt rows 0..255, img rows 256..2303) ----
  const _Float16* Xi = X_h + (size_t)TXTN * DIMN;
  dim3 gT(DIMN / 128, TXTN / 64), gI(DIMN / 128, IMGN / 64);
  gemm_wmma_tdm<<<gT, 256, 0, stream>>>(X_h, Waq_h, baq, qf,                       TXTN, DIMN, DIMN);
  gemm_wmma_tdm<<<gI, 256, 0, stream>>>(Xi,  Wq_h,  bq,  qf + (size_t)TXTN * DIMN, IMGN, DIMN, DIMN);
  gemm_wmma_tdm<<<gT, 256, 0, stream>>>(X_h, Wak_h, bak, kf,                       TXTN, DIMN, DIMN);
  gemm_wmma_tdm<<<gI, 256, 0, stream>>>(Xi,  Wk_h,  bk,  kf + (size_t)TXTN * DIMN, IMGN, DIMN, DIMN);
  gemm_wmma_tdm<<<gT, 256, 0, stream>>>(X_h, Wav_h, bav, vf,                       TXTN, DIMN, DIMN);
  gemm_wmma_tdm<<<gI, 256, 0, stream>>>(Xi,  Wv_h,  bv,  vf + (size_t)TXTN * DIMN, IMGN, DIMN, DIMN);

  // ---- RoPE + pack ----
  rope_pack<<<(SEQN * (DIMN / 2) + 255) / 256, 256, 0, stream>>>(
      qf, kf, img_cos, img_sin, txt_cos, txt_sin, qh, kh);
  pack_vT<<<(DIMN * SEQN + 255) / 256, 256, 0, stream>>>(vf, vT);

  // ---- attention ----
  attn_wmma<<<dim3(NH, SEQN / 128), 256, 0, stream>>>(qh, kh, vT, ao);

  // ---- output projections straight into d_out (img first, then txt) ----
  gemm_wmma_tdm<<<gI, 256, 0, stream>>>(ao + (size_t)TXTN * DIMN, Wo_h, bo, out, IMGN, DIMN, DIMN);
  gemm_wmma_tdm<<<gT, 256, 0, stream>>>(ao, Wao_h, bao, out + (size_t)IMGN * DIMN, TXTN, DIMN, DIMN);
}
